// CausalVideoAttention_81870666597062
// MI455X (gfx1250) — matrix operand order repaired and checked by reading, not compiled, not run on hardware
//
#include <hip/hip_runtime.h>
#include <hip/hip_bf16.h>

// ---------------------------------------------------------------------------
// CausalVideoAttention for MI455X (gfx1250, wave32, WMMA 16x16x32 f16->f32)
//
//   1) cast f32 -> f16 for x, W_qkv, W_o
//   2) QKV GEMM (double-buffered LDS, async global->LDS when available)
//   3) RMSNorm over 64-elem head vectors of q,k (f32 math)
//   4) Flash attention, 64-key blocks, DPP ROW_XMASK softmax reductions,
//      V staged transposed in LDS so PV B-fragments are contiguous b128 reads
//   5) Output GEMM -> f32 d_out
//
// Workspace: [0,16M) xh | [16M,22M) wqkvh | [22M,24M) woh |
//            [24M,72M) qkvh | [72M,88M) zh   (needs ws_size >= 88 MB)
// ---------------------------------------------------------------------------

#define B_    4
#define S_    2048
#define D_    1024
#define H_    16
#define DH_   64
#define ROWS_ (B_ * S_)     // 8192
#define QKVW_ (3 * D_)      // 3072

typedef __attribute__((ext_vector_type(16))) _Float16 v16h;
typedef __attribute__((ext_vector_type(8)))  _Float16 v8h;
typedef __attribute__((ext_vector_type(8)))  float    v8f;

// --- optional gfx1250 async global->LDS copy path ---------------------------
#if defined(__has_builtin)
#  if __has_builtin(__builtin_amdgcn_global_load_async_to_lds_b128)
#    define USE_ASYNC_COPY 1
#  endif
#endif
#ifndef USE_ASYNC_COPY
#  define USE_ASYNC_COPY 0
#endif

#if USE_ASYNC_COPY
typedef int v4i_ __attribute__((vector_size(16)));
typedef __attribute__((address_space(1))) v4i_* gptr_v4i;   // global b128
typedef __attribute__((address_space(3))) v4i_* lptr_v4i;   // LDS b128
#define ASYNC_B128(gp, lp) \
  __builtin_amdgcn_global_load_async_to_lds_b128((gptr_v4i)(gp), (lptr_v4i)(lp), 0, 0)
#define WAIT_ASYNC() asm volatile("s_wait_asynccnt 0" ::: "memory")
#endif

// --- DPP 16-lane butterfly reductions (ROW_XMASK, pure VALU) ----------------
template <int CTRL>
__device__ __forceinline__ float dpp_movf(float x) {
  return __builtin_bit_cast(
      float, __builtin_amdgcn_update_dpp(0, __builtin_bit_cast(int, x), CTRL,
                                         0xf, 0xf, true));
}
__device__ __forceinline__ float row16_max(float x) {
  x = fmaxf(x, dpp_movf<0x161>(x));   // ROW_XMASK:1
  x = fmaxf(x, dpp_movf<0x162>(x));   // ROW_XMASK:2
  x = fmaxf(x, dpp_movf<0x164>(x));   // ROW_XMASK:4
  x = fmaxf(x, dpp_movf<0x168>(x));   // ROW_XMASK:8
  return x;
}
__device__ __forceinline__ float row16_sum(float x) {
  x += dpp_movf<0x161>(x);
  x += dpp_movf<0x162>(x);
  x += dpp_movf<0x164>(x);
  x += dpp_movf<0x168>(x);
  return x;
}

// ---------------------------------------------------------------------------
// 1) elementwise f32 -> f16 cast
// ---------------------------------------------------------------------------
__global__ void __launch_bounds__(256) cast_f32_f16(const float* __restrict__ in,
                                                    _Float16* __restrict__ out, int n) {
  int idx = (blockIdx.x * blockDim.x + threadIdx.x) * 4;
  if (idx + 4 <= n) {
    float4 v = *(const float4*)(in + idx);
    out[idx + 0] = (_Float16)v.x;
    out[idx + 1] = (_Float16)v.y;
    out[idx + 2] = (_Float16)v.z;
    out[idx + 3] = (_Float16)v.w;
  } else {
    for (int i = idx; i < n; ++i) out[i] = (_Float16)in[i];
  }
}

// ---------------------------------------------------------------------------
// 2/5) f16 GEMM, f32 accum: C[M,N] = A[M,K] @ B[K,N] + bias
// Block tile 128x128, 8 waves; wave = full-height 128x16 strip (8 accums).
// K-tile 32, double-buffered LDS, padded strides (A:40, B:136 halves).
// ---------------------------------------------------------------------------
#define ASTR 40
#define BSTR 136

template <bool OUT_F16>
__global__ void __launch_bounds__(256)
gemm_f16_wmma(const _Float16* __restrict__ A, const _Float16* __restrict__ Bm,
              const float* __restrict__ bias, void* __restrict__ Cout,
              int M, int N, int K) {
  __shared__ _Float16 lds_a[2][128 * ASTR];   // 2 x 10.0 KB
  __shared__ _Float16 lds_b[2][32 * BSTR];    // 2 x  8.5 KB

  const int tid  = threadIdx.x;
  const int lane = tid & 31;
  const int wave = tid >> 5;               // 0..7
  const int m0   = blockIdx.y * 128;
  const int n0   = blockIdx.x * 128;
  const int wn   = wave * 16;              // wave's 16-col strip

  v8f acc[8];
#pragma unroll
  for (int mi = 0; mi < 8; ++mi)
#pragma unroll
    for (int j = 0; j < 8; ++j) acc[mi][j] = 0.0f;

  // cooperative staging: 1 b128 per thread per matrix per K-tile
  const int sa_row = tid >> 1, sa_col = (tid & 1) * 16;   // A: 128x32
  const int sb_row = tid >> 3, sb_col = (tid & 7) * 16;   // B: 32x128
  const _Float16* gA = A + (size_t)(m0 + sa_row) * K + sa_col;
  const _Float16* gB = Bm + (size_t)sb_row * N + n0 + sb_col;
  const int la_off = sa_row * ASTR + sa_col;
  const int lb_off = sb_row * BSTR + sb_col;

  // fragment indices (wave32, contiguous k = 16*(lane>>4) + i in both A and B)
  const int a_row = lane & 15;
  const int a_k0  = (lane >> 4) * 16;
  const int b_col = lane & 15;
  const int b_k0  = (lane >> 4) * 16;

  const int NT = K / 32;

  // prologue: tile 0 -> buffer 0
#if USE_ASYNC_COPY
  ASYNC_B128(gA, &lds_a[0][la_off]);
  ASYNC_B128(gB, &lds_b[0][lb_off]);
  WAIT_ASYNC();
#else
  *(v16h*)&lds_a[0][la_off] = *(const v16h*)gA;
  *(v16h*)&lds_b[0][lb_off] = *(const v16h*)gB;
#endif
  __syncthreads();

  for (int t = 0; t < NT; ++t) {
    const int cur = t & 1, nxt = cur ^ 1;
    const bool more = (t + 1 < NT);

#if USE_ASYNC_COPY
    if (more) {   // async copies overlap with the WMMAs below
      ASYNC_B128(gA + (t + 1) * 32, &lds_a[nxt][la_off]);
      ASYNC_B128(gB + (size_t)(t + 1) * 32 * N, &lds_b[nxt][lb_off]);
    }
#else
    v16h ra = {}, rb = {};
    if (more) {   // global loads overlap with the WMMAs below
      ra = *(const v16h*)(gA + (t + 1) * 32);
      rb = *(const v16h*)(gB + (size_t)(t + 1) * 32 * N);
    }
    if (t + 2 < NT) {
      __builtin_prefetch(gA + (t + 2) * 32, 0, 0);
      __builtin_prefetch(gB + (size_t)(t + 2) * 32 * N, 0, 0);
    }
#endif

    // one B fragment per wave per K-tile (16-col strip)
    v16h bf;
#pragma unroll
    for (int i = 0; i < 16; ++i)
      bf[i] = lds_b[cur][(b_k0 + i) * BSTR + wn + b_col];

#pragma unroll
    for (int mi = 0; mi < 8; ++mi) {
      const _Float16* p = &lds_a[cur][(mi * 16 + a_row) * ASTR + a_k0];
      v8h lo = *(const v8h*)p;
      v8h hi = *(const v8h*)(p + 8);
      v16h af;
#pragma unroll
      for (int i = 0; i < 8; ++i) { af[i] = lo[i]; af[8 + i] = hi[i]; }
      acc[mi] = __builtin_amdgcn_wmma_f32_16x16x32_f16(
          false, af, false, bf, (short)0, acc[mi], false, false);
    }

    if (more) {
#if USE_ASYNC_COPY
      WAIT_ASYNC();
#else
      *(v16h*)&lds_a[nxt][la_off] = ra;
      *(v16h*)&lds_b[nxt][lb_off] = rb;
#endif
      __syncthreads();
    }
  }

  // epilogue
  const int crow = (lane >> 4) * 8;
  const int ccol = lane & 15;
  const int gcol = n0 + wn + ccol;
  const float bv = bias[gcol];
#pragma unroll
  for (int mi = 0; mi < 8; ++mi)
#pragma unroll
    for (int i = 0; i < 8; ++i) {
      const int grow = m0 + mi * 16 + crow + i;
      const float v = acc[mi][i] + bv;
      if (OUT_F16)
        ((_Float16*)Cout)[(size_t)grow * N + gcol] = (_Float16)v;
      else
        ((float*)Cout)[(size_t)grow * N + gcol] = v;
    }
}

// ---------------------------------------------------------------------------
// 3) in-place RMSNorm over 64-element head vectors of q and k
// ---------------------------------------------------------------------------
__global__ void __launch_bounds__(256)
rmsnorm_qk(_Float16* __restrict__ qkv, const float* __restrict__ wq,
           const float* __restrict__ wk) {
  const int wave = threadIdx.x >> 5;
  const int lane = threadIdx.x & 31;
  const long g   = (long)blockIdx.x * 8 + wave;
  const int row  = (int)(g >> 5);
  const int sub  = (int)(g & 31);
  const int qk   = sub >> 4;
  const int h    = sub & 15;
  _Float16* p = qkv + (size_t)row * QKVW_ + qk * D_ + h * DH_ + lane * 2;
  float x0 = (float)p[0], x1 = (float)p[1];
  float ss = x0 * x0 + x1 * x1;
#pragma unroll
  for (int m = 16; m >= 1; m >>= 1) ss += __shfl_xor(ss, m, 32);
  const float scale = rsqrtf(ss * (1.0f / 64.0f) + 1e-6f);
  const float* w = qk ? wk : wq;
  p[0] = (_Float16)(x0 * scale * w[lane * 2 + 0]);
  p[1] = (_Float16)(x1 * scale * w[lane * 2 + 1]);
}

// ---------------------------------------------------------------------------
// 4) flash attention: grid (S/64, B*H), 4 waves x 16 q-rows, 64-key blocks.
//    8 WMMA scores + 8 WMMA PV per block; DPP softmax reductions.
//    V staged TRANSPOSED (lds_vt[d][key], stride 72 halves) so each PV
//    B-fragment is one contiguous 32-byte LDS read (2x ds_load_b128).
// ---------------------------------------------------------------------------
#define VSTR 72

__global__ void __launch_bounds__(128)
flash_attn(const _Float16* __restrict__ qkv, _Float16* __restrict__ z) {
  __shared__ _Float16 lds_vt[64 * VSTR];     // 9 KB transposed V tile
  __shared__ _Float16 lds_p[4][16 * 64];     // 8 KB P staging (per wave)

  const int tid  = threadIdx.x;
  const int lane = tid & 31;
  const int wave = tid >> 5;
  const int qblk = blockIdx.x;
  const int bh   = blockIdx.y;
  const int b    = bh >> 4;
  const int h    = bh & 15;

  const _Float16* qbase = qkv + (size_t)b * S_ * QKVW_ + h * DH_;
  const _Float16* kbase = qbase + D_;
  const _Float16* vbase = qbase + 2 * D_;

  const int q0 = qblk * 64 + wave * 16;

  const int a_row = lane & 15, a_k0 = (lane >> 4) * 16;
  const int b_col = lane & 15, b_k0 = (lane >> 4) * 16;
  const int crow  = (lane >> 4) * 8, ccol = lane & 15;

  // preload Q fragments (16x64 -> two 16x32 A-frags)
  v16h qf[2];
#pragma unroll
  for (int d = 0; d < 2; ++d) {
    const _Float16* p = qbase + (size_t)(q0 + a_row) * QKVW_ + d * 32 + a_k0;
#pragma unroll
    for (int i = 0; i < 8; ++i) { qf[d][i] = p[i]; qf[d][8 + i] = p[8 + i]; }
  }

  float mrow[8], lrow[8];
  v8f o[4];
#pragma unroll
  for (int i = 0; i < 8; ++i) { mrow[i] = -__builtin_inff(); lrow[i] = 0.0f; }
#pragma unroll
  for (int nt = 0; nt < 4; ++nt)
#pragma unroll
    for (int j = 0; j < 8; ++j) o[nt][j] = 0.0f;

  const int nkb = qblk + 1;                  // causal: keys < (qblk+1)*64
  const int vr = tid >> 1, vc = (tid & 1) * 16;   // thread's V row / col base

  for (int kb = 0; kb < nkb; ++kb) {
    const int k0 = kb * 64;

    // stage V tile transposed: lds_vt[d * VSTR + key]
    {
      const _Float16* vp = vbase + (size_t)(k0 + vr) * QKVW_ + vc;
      v16h v0 = *(const v16h*)vp;
      v16h v1 = *(const v16h*)(vp + 32);
#pragma unroll
      for (int j = 0; j < 16; ++j) {
        lds_vt[(vc + j) * VSTR + vr]      = v0[j];
        lds_vt[(vc + 32 + j) * VSTR + vr] = v1[j];
      }
      if (kb + 1 < nkb) {   // prefetch next key-block's V rows
        __builtin_prefetch(vp + (size_t)64 * QKVW_, 0, 0);
        __builtin_prefetch(vp + (size_t)64 * QKVW_ + 32, 0, 0);
      }
    }
    __syncthreads();

    // scores S = Q @ K^T : 4 N-tiles x 2 k-steps (K rows contiguous in mem)
    v8f s[4];
#pragma unroll
    for (int nt = 0; nt < 4; ++nt) {
#pragma unroll
      for (int j = 0; j < 8; ++j) s[nt][j] = 0.0f;
#pragma unroll
      for (int d = 0; d < 2; ++d) {
        const _Float16* p =
            kbase + (size_t)(k0 + nt * 16 + b_col) * QKVW_ + d * 32 + b_k0;
        v16h bf;
#pragma unroll
        for (int i = 0; i < 16; ++i) bf[i] = p[i];
        s[nt] = __builtin_amdgcn_wmma_f32_16x16x32_f16(
            false, qf[d], false, bf, (short)0, s[nt], false, false);
      }
    }

    // causal mask only where this wave's rows can be masked (uniform branch)
    const bool need_mask = (k0 + 63 > q0);

#pragma unroll
    for (int i = 0; i < 8; ++i) {
      const int qpos = q0 + crow + i;
      if (need_mask) {
#pragma unroll
        for (int nt = 0; nt < 4; ++nt)
          if (k0 + nt * 16 + ccol > qpos) s[nt][i] = -__builtin_inff();
      }
      float m = fmaxf(fmaxf(s[0][i], s[1][i]), fmaxf(s[2][i], s[3][i]));
      m = row16_max(m);                                   // DPP reduction
      const float mnew = fmaxf(mrow[i], m);
      const float corr = __expf(mrow[i] - mnew);
      lrow[i] *= corr;
#pragma unroll
      for (int nt = 0; nt < 4; ++nt) o[nt][i] *= corr;
      float ps = 0.0f;
#pragma unroll
      for (int nt = 0; nt < 4; ++nt) {
        const float pv = __expf(s[nt][i] - mnew);
        lds_p[wave][(crow + i) * 64 + nt * 16 + ccol] = (_Float16)pv;
        ps += pv;
      }
      lrow[i] += row16_sum(ps);                           // DPP reduction
      mrow[i]  = mnew;
    }

    // same-wave LDS RAW before re-reading P as A-fragments
    asm volatile("s_wait_dscnt 0" ::: "memory");

    v16h pf[2];
#pragma unroll
    for (int d2 = 0; d2 < 2; ++d2) {
      const _Float16* pp = &lds_p[wave][a_row * 64 + d2 * 32 + a_k0];
#pragma unroll
      for (int i = 0; i < 8; ++i) { pf[d2][i] = pp[i]; pf[d2][8 + i] = pp[8 + i]; }
    }

    // O += P @ V : 4 d-tiles x 2 k-steps; V frags are contiguous b128 reads
#pragma unroll
    for (int nt = 0; nt < 4; ++nt)
#pragma unroll
      for (int d2 = 0; d2 < 2; ++d2) {
        const v16h vf =
            *(const v16h*)&lds_vt[(nt * 16 + b_col) * VSTR + d2 * 32 + b_k0];
        o[nt] = __builtin_amdgcn_wmma_f32_16x16x32_f16(
            false, pf[d2], false, vf, (short)0, o[nt], false, false);
      }
    __syncthreads();   // protect lds_vt before next block's staging
  }

  // finalize: divide by row sums, scatter into z[b,s,h,d] (f16)
#pragma unroll
  for (int nt = 0; nt < 4; ++nt)
#pragma unroll
    for (int i = 0; i < 8; ++i) {
      const int qpos = q0 + crow + i;
      z[((size_t)b * S_ + qpos) * D_ + h * DH_ + nt * 16 + ccol] =
          (_Float16)(o[nt][i] / lrow[i]);
    }
}

// ---------------------------------------------------------------------------
// launcher
// ---------------------------------------------------------------------------
extern "C" void kernel_launch(void* const* d_in, const int* in_sizes, int n_in,
                              void* d_out, int out_size, void* d_ws, size_t ws_size,
                              hipStream_t stream) {
  const float* x     = (const float*)d_in[0];
  // d_in[1] = mask (causality computed analytically; unused)
  const float* W_qkv = (const float*)d_in[2];
  const float* b_qkv = (const float*)d_in[3];
  const float* W_o   = (const float*)d_in[4];
  const float* b_o   = (const float*)d_in[5];
  const float* wq    = (const float*)d_in[6];
  const float* wk    = (const float*)d_in[7];
  float* out = (float*)d_out;

  char* ws = (char*)d_ws;
  _Float16* xh    = (_Float16*)(ws);                 // 16 MB
  _Float16* wqkvh = (_Float16*)(ws + (16ull << 20)); //  6 MB
  _Float16* woh   = (_Float16*)(ws + (22ull << 20)); //  2 MB
  _Float16* qkvh  = (_Float16*)(ws + (24ull << 20)); // 48 MB
  _Float16* zh    = (_Float16*)(ws + (72ull << 20)); // 16 MB

  {
    int n = ROWS_ * D_;
    cast_f32_f16<<<(n / 4 + 255) / 256, 256, 0, stream>>>(x, xh, n);
    n = D_ * QKVW_;
    cast_f32_f16<<<(n / 4 + 255) / 256, 256, 0, stream>>>(W_qkv, wqkvh, n);
    n = D_ * D_;
    cast_f32_f16<<<(n / 4 + 255) / 256, 256, 0, stream>>>(W_o, woh, n);
  }

  gemm_f16_wmma<true><<<dim3(QKVW_ / 128, ROWS_ / 128), 256, 0, stream>>>(
      xh, wqkvh, b_qkv, (void*)qkvh, ROWS_, QKVW_, D_);

  rmsnorm_qk<<<(ROWS_ * 32) / 8, 256, 0, stream>>>(qkvh, wq, wk);

  flash_attn<<<dim3(S_ / 64, B_ * H_), 128, 0, stream>>>(qkvh, zh);

  gemm_f16_wmma<false><<<dim3(D_ / 128, ROWS_ / 128), 256, 0, stream>>>(
      zh, woh, b_o, (void*)out, ROWS_, D_, D_);
}
